// Network_39307540693901
// MI455X (gfx1250) — matrix operand order repaired and checked by reading, not compiled, not run on hardware
//
#include <hip/hip_runtime.h>

typedef __attribute__((ext_vector_type(16))) _Float16 v16h;
typedef __attribute__((ext_vector_type(8)))  _Float16 v8h;
typedef __attribute__((ext_vector_type(8)))  float    v8f;

#define T_BINS   350
#define C1       410
#define C1P      416     // 26 tiles of 16
#define C2       10
#define C2P      16
#define KIN      3072
#define BATCH    64
#define THETA_F  10.0f

// ---------------------------------------------------------------------------
// Kernel 1: a1[64 x 416] = x[64 x 3072] * W1^T   (W1 is [410 x 3072] row-major,
// so W1 rows are contiguous in K == exactly the WMMA B-operand K-runs).
// One wave per 16x16 tile; K loop of 96 x v_wmma_f32_16x16x32_f16.
//
// Pad columns (n >= 410) alias W1 row 0 unmasked: B column n only contributes
// to C column n, pad C columns feed pad s1T columns, and GEMM2's B operand is
// zero-guarded for k >= 410 -- so the pad garbage is algebraically dead.
// ---------------------------------------------------------------------------
__global__ __launch_bounds__(32) void gemm1_wmma(
    const float* __restrict__ x, const float* __restrict__ W1,
    float* __restrict__ a1out)
{
  const int tile = blockIdx.x;            // 104 tiles = 4 (M) x 26 (N)
  const int tm   = tile / 26;
  const int tn   = tile % 26;
  const int lane = threadIdx.x;
  const int half = lane >> 4;
  const int l16  = lane & 15;
  const int brow = tm * 16 + l16;         // A row (batch index), always < 64
  const int ncol = tn * 16 + l16;         // B column (out channel), may be >= 410
  const float* __restrict__ wrow = W1 + (size_t)(ncol < C1 ? ncol : 0) * KIN;
  const float* __restrict__ xrow = x + (size_t)brow * KIN;

  v8f acc = {};
  #pragma unroll 2
  for (int kb = 0; kb < KIN; kb += 32) {
    // --- A operand: lane needs K = [kA, kA+15], kA = kb + half*16
    const int kA = kb + half * 16;
    const float4 a0 = *(const float4*)(xrow + kA);
    const float4 a1v = *(const float4*)(xrow + kA + 4);
    const float4 a2v = *(const float4*)(xrow + kA + 8);
    const float4 a3v = *(const float4*)(xrow + kA + 12);
    // --- B operand: lane needs K = [kB, kB+15], kB = kb + half*16 (contiguous W1 row)
    const int kB = kb + half * 16;
    const float4 b0 = *(const float4*)(wrow + kB);
    const float4 b1 = *(const float4*)(wrow + kB + 4);
    const float4 b2 = *(const float4*)(wrow + kB + 8);
    const float4 b3 = *(const float4*)(wrow + kB + 12);
    // Prefetch next K block into cache while this iteration's WMMAs run
    // (global_prefetch_b8; speculative, OOB on last iteration is dropped).
    __builtin_prefetch(xrow + kA + 32, 0, 3);
    __builtin_prefetch(wrow + kB + 32, 0, 3);

    v16h A;
    A[0]=(_Float16)a0.x;  A[1]=(_Float16)a0.y;  A[2]=(_Float16)a0.z;  A[3]=(_Float16)a0.w;
    A[4]=(_Float16)a1v.x; A[5]=(_Float16)a1v.y; A[6]=(_Float16)a1v.z; A[7]=(_Float16)a1v.w;
    A[8]=(_Float16)a2v.x; A[9]=(_Float16)a2v.y; A[10]=(_Float16)a2v.z; A[11]=(_Float16)a2v.w;
    A[12]=(_Float16)a3v.x;A[13]=(_Float16)a3v.y;A[14]=(_Float16)a3v.z; A[15]=(_Float16)a3v.w;
    v16h B;
    B[0]=(_Float16)b0.x;  B[1]=(_Float16)b0.y;  B[2]=(_Float16)b0.z;  B[3]=(_Float16)b0.w;
    B[4]=(_Float16)b1.x;  B[5]=(_Float16)b1.y;  B[6]=(_Float16)b1.z;  B[7]=(_Float16)b1.w;
    B[8]=(_Float16)b2.x;  B[9]=(_Float16)b2.y;  B[10]=(_Float16)b2.z; B[11]=(_Float16)b2.w;
    B[12]=(_Float16)b3.x; B[13]=(_Float16)b3.y; B[14]=(_Float16)b3.z; B[15]=(_Float16)b3.w;

    acc = __builtin_amdgcn_wmma_f32_16x16x32_f16(false, A, false, B,
                                                 (short)0, acc, false, false);
  }
  // C/D layout: VGPR v holds M = v + half*8, N = l16
  #pragma unroll
  for (int v = 0; v < 8; ++v) {
    const int m = v + half * 8;
    a1out[(size_t)(tm * 16 + m) * C1P + tn * 16 + l16] = acc[v];
  }
}

// ---------------------------------------------------------------------------
// Kernel 2: layer-1 spike scan. psp(a1) on a time-constant drive == a1 * S(t);
// S(t) generated by the exact alpha-kernel 2-state recurrence with unit input.
// Refractory = 10-deep shift buffer exactly as the reference. One lane/neuron.
// Writes s1^T as f16 [B*T, C1P] (K-contiguous for GEMM2's A operand).
// ---------------------------------------------------------------------------
__global__ void spike1_scan(const float* __restrict__ a1,
                            _Float16* __restrict__ s1T)
{
  const int idx = blockIdx.x * blockDim.x + threadIdx.x;
  if (idx >= BATCH * C1P) return;
  const int c = idx % C1P;
  const int b = idx / C1P;
  const float av = a1[(size_t)b * C1P + c];

  const float r  = __expf(-0.1f);             // e^{-Ts/tauSr}
  const float ga = __expf(1.0f) * 0.1f;       // e/tauSr
  float nu[10];
  #pragma unroll
  for (int j = 0; j < 10; ++j)
    nu[j] = -2.0f * THETA_F * (float)(j + 1) * __expf(1.0f - (float)(j + 1));

  float rbuf[10];
  #pragma unroll
  for (int j = 0; j < 10; ++j) rbuf[j] = 0.0f;

  float Ap = 0.0f, Bp = 0.0f;
  for (int t = 0; t < T_BINS; ++t) {
    const float nA = 1.0f + r * Ap;           // unit-input alpha recurrence
    const float nB = r * (Ap + Bp);
    Ap = nA; Bp = nB;
    const float ue = ga * nB * av + rbuf[0];
    const float s  = (ue >= THETA_F) ? 1.0f : 0.0f;
    #pragma unroll
    for (int j = 0; j < 9; ++j) rbuf[j] = rbuf[j + 1] + s * nu[j];
    rbuf[9] = s * nu[9];
    s1T[(size_t)(b * T_BINS + t) * C1P + c] = (_Float16)s;
  }
}

// ---------------------------------------------------------------------------
// Kernel 3: a2[(B*T) x 16] = s1^T[(B*T) x 416] * W2^T.  M = 22400 (1400 tiles),
// N = 16 (10 real), K = 416 (410 real) -> 13 x v_wmma per wave.
// ---------------------------------------------------------------------------
__global__ __launch_bounds__(32) void gemm2_wmma(
    const _Float16* __restrict__ s1T, const float* __restrict__ W2,
    float* __restrict__ a2)
{
  const int tile = blockIdx.x;               // 0..1399 (M tiles)
  const int lane = threadIdx.x;
  const int half = lane >> 4;
  const int l16  = lane & 15;
  const int mrow = tile * 16 + l16;          // < 22400 always
  const _Float16* __restrict__ arow = s1T + (size_t)mrow * C1P;

  v8f acc = {};
  #pragma unroll 2
  for (int kb = 0; kb < C1P; kb += 32) {
    const int kA = kb + half * 8;
    const v8h lo = *(const v8h*)(arow + kA);        // 16B load of 8 halfs
    const v8h hi = *(const v8h*)(arow + kA + 16);
    __builtin_prefetch(arow + kA + 32, 0, 3);       // next K block
    v16h A;
    #pragma unroll
    for (int j = 0; j < 8; ++j) { A[j] = lo[j]; A[j + 8] = hi[j]; }

    v16h B;                                         // W2 is tiny & L2-resident
    #pragma unroll
    for (int j = 0; j < 16; ++j) {
      const int k = kb + half * 8 + (j < 8 ? j : j + 8);
      const float w = (l16 < C2 && k < C1) ? W2[(size_t)l16 * C1 + k] : 0.0f;
      B[j] = (_Float16)w;
    }
    acc = __builtin_amdgcn_wmma_f32_16x16x32_f16(false, A, false, B,
                                                 (short)0, acc, false, false);
  }
  #pragma unroll
  for (int v = 0; v < 8; ++v) {
    const int m = v + half * 8;
    a2[(size_t)(tile * 16 + m) * C2P + l16] = acc[v];
  }
}

// ---------------------------------------------------------------------------
// Kernel 4: layer-2 psp (exact alpha-kernel IIR recurrence) + spike scan.
// One lane per (b, o) neuron; writes s2 [64, 10, 350] f32 to d_out.
// ---------------------------------------------------------------------------
__global__ void spike2_scan(const float* __restrict__ a2,
                            float* __restrict__ out)
{
  const int idx = blockIdx.x * blockDim.x + threadIdx.x;
  if (idx >= BATCH * C2) return;
  const int o = idx % C2;
  const int b = idx / C2;

  const float r  = __expf(-0.1f);
  const float ga = __expf(1.0f) * 0.1f;
  float nu[10];
  #pragma unroll
  for (int j = 0; j < 10; ++j)
    nu[j] = -2.0f * THETA_F * (float)(j + 1) * __expf(1.0f - (float)(j + 1));

  float rbuf[10];
  #pragma unroll
  for (int j = 0; j < 10; ++j) rbuf[j] = 0.0f;

  float Ap = 0.0f, Bp = 0.0f;
  for (int t = 0; t < T_BINS; ++t) {
    const float xv = a2[(size_t)(b * T_BINS + t) * C2P + o];
    const float nA = xv + r * Ap;
    const float nB = r * (Ap + Bp);
    Ap = nA; Bp = nB;
    const float ue = ga * nB + rbuf[0];
    const float s  = (ue >= THETA_F) ? 1.0f : 0.0f;
    #pragma unroll
    for (int j = 0; j < 9; ++j) rbuf[j] = rbuf[j + 1] + s * nu[j];
    rbuf[9] = s * nu[9];
    out[(size_t)(b * C2 + o) * T_BINS + t] = s;
  }
}

// ---------------------------------------------------------------------------
extern "C" void kernel_launch(void* const* d_in, const int* in_sizes, int n_in,
                              void* d_out, int out_size, void* d_ws, size_t ws_size,
                              hipStream_t stream) {
  const float* x  = (const float*)d_in[0];   // [64,3,32,32] == [64,3072]
  const float* W1 = (const float*)d_in[1];   // [410,3072]
  const float* W2 = (const float*)d_in[2];   // [10,410]
  float* out = (float*)d_out;                // [64,10,350]

  char* ws = (char*)d_ws;
  const size_t A1_BYTES  = (size_t)BATCH * C1P * sizeof(float);          // 106496
  const size_t S1T_BYTES = (size_t)BATCH * T_BINS * C1P * sizeof(short); // ~18.6 MB
  float*    a1  = (float*)ws;
  _Float16* s1T = (_Float16*)(ws + A1_BYTES);
  float*    a2  = (float*)(ws + A1_BYTES + S1T_BYTES);                   // ~1.4 MB

  // fc1: 4x26 = 104 output tiles, one wave each
  gemm1_wmma<<<dim3(104), dim3(32), 0, stream>>>(x, W1, a1);
  // layer-1 scan: 64*416 neurons (pads feed zero-weighted K columns in GEMM2)
  spike1_scan<<<dim3((BATCH * C1P + 255) / 256), dim3(256), 0, stream>>>(a1, s1T);
  // fc2: 1400 M-tiles, one wave each
  gemm2_wmma<<<dim3(1400), dim3(32), 0, stream>>>(s1T, W2, a2);
  // layer-2 psp + scan: 640 neurons
  spike2_scan<<<dim3((BATCH * C2 + 63) / 64), dim3(64), 0, stream>>>(a2, out);
}
